// LSTM_17892833755638
// MI455X (gfx1250) — matrix operand order repaired and checked by reading, not compiled, not run on hardware
//
#include <hip/hip_runtime.h>
#include <cstddef>

// Problem constants (match reference)
#define SEQ    512
#define BATCH  32
#define INP    1024
#define HID    1024
#define LAYERS 4
#define G4     (4 * HID)   // 4096 gate width

// ---------- types for WMMA fragments (wave32, gfx1250) ----------
typedef __bf16 bf16_t;
typedef __attribute__((ext_vector_type(8)))  bf16_t v8bf;
typedef __attribute__((ext_vector_type(16))) bf16_t v16bf;
typedef __attribute__((ext_vector_type(8)))  float  v8f;

__device__ __forceinline__ unsigned short f2bf(float f) {
  // round-to-nearest-even fp32 -> bf16 (bits)
  unsigned int x = __float_as_uint(f);
  x += 0x7FFFu + ((x >> 16) & 1u);
  return (unsigned short)(x >> 16);
}

__device__ __forceinline__ float sigf(float x) {
  return 1.0f / (1.0f + __expf(-x));
}

// Two 16-byte loads -> one 32-byte bf16 fragment register group
__device__ __forceinline__ v16bf load_frag(const unsigned short* p0,
                                           const unsigned short* p1) {
  union { v16bf w; v8bf h[2]; } u;
  u.h[0] = *(const v8bf*)p0;
  u.h[1] = *(const v8bf*)p1;
  return u.w;
}

// ---------------------------------------------------------------
// fp32 -> bf16 bulk convert (weights, input activations)
__global__ void f32_to_bf16_kernel(const float* __restrict__ in,
                                   unsigned short* __restrict__ out,
                                   size_t n) {
  size_t i = (size_t)blockIdx.x * blockDim.x + threadIdx.x;
  size_t stride = (size_t)gridDim.x * blockDim.x;
  for (; i < n; i += stride) out[i] = f2bf(in[i]);
}

// bias = b_ih + b_hh (fp32), all layers at once
__global__ void bias_combine_kernel(const float* __restrict__ bih,
                                    const float* __restrict__ bhh,
                                    float* __restrict__ bias, int n) {
  int i = blockIdx.x * blockDim.x + threadIdx.x;
  if (i < n) bias[i] = bih[i] + bhh[i];
}

// zero h (both bf16 ping-pong buffers), h_f32, c
__global__ void zero_state_kernel(unsigned short* __restrict__ h0,
                                  unsigned short* __restrict__ h1,
                                  float* __restrict__ hf,
                                  float* __restrict__ c, int n) {
  int i = blockIdx.x * blockDim.x + threadIdx.x;
  if (i < n) { h0[i] = 0; h1[i] = 0; hf[i] = 0.0f; c[i] = 0.0f; }
}

// copy fp32 final h/c into d_out hs/cs slots
__global__ void copy_states_kernel(const float* __restrict__ hf,
                                   const float* __restrict__ cf,
                                   float* __restrict__ out_h,
                                   float* __restrict__ out_c, int n) {
  int i = blockIdx.x * blockDim.x + threadIdx.x;
  if (i < n) { out_h[i] = hf[i]; out_c[i] = cf[i]; }
}

// ---------------------------------------------------------------
// Big parallel GEMM: XG[row, g] = bias[g] + sum_d Xbf[row, d] * Wih[g, d]
// rows = SEQ*BATCH = 16384, g in [0, 4H), K = INP.
// One wave computes a 64(M) x 64(N) register tile: 4x4 accumulator tiles.
// B fragments are loaded once per k-step and reused across 4 A fragments
// (~32 FLOP per L2 byte; W_ih re-streamed only 256x instead of 1024x).
// __launch_bounds__(256, 1): ~210 live VGPRs/lane; allow full register file
// (L2-bound kernel, one workgroup per SIMD is sufficient) -> no spills.
__global__ __launch_bounds__(256, 1)
void xgate_gemm_kernel(const unsigned short* __restrict__ Xbf,  // [S*B, INP] bf16
                       const unsigned short* __restrict__ Wih,  // [4H, INP] bf16
                       const float* __restrict__ bias,          // [4H]
                       float* __restrict__ XG) {                // [S*B, 4H] f32
  const int lane = threadIdx.x & 31;
  const int wave = blockIdx.x * (blockDim.x >> 5) + (threadIdx.x >> 5);
  const int NG   = G4 / 64;             // 64 n-groups of width 64
  const int m_block = (wave / NG) * 64; // 256 m-blocks of 64 rows
  const int ngrp    = wave % NG;
  const int lm = lane & 15;
  const int lh = lane >> 4;             // 0 or 1

  v8f acc[4][4];
#pragma unroll
  for (int s = 0; s < 4; ++s) {
    const float bv = bias[ngrp * 64 + s * 16 + lm];
#pragma unroll
    for (int mi = 0; mi < 4; ++mi)
#pragma unroll
      for (int r = 0; r < 8; ++r) acc[mi][s][r] = bv;
  }

  const unsigned short* arow[4];
#pragma unroll
  for (int mi = 0; mi < 4; ++mi)
    arow[mi] = Xbf + (size_t)(m_block + mi * 16 + lm) * INP;
  const unsigned short* brow[4];
#pragma unroll
  for (int s = 0; s < 4; ++s)
    brow[s] = Wih + (size_t)(ngrp * 64 + s * 16 + lm) * INP + lh * 16;

  for (int k0 = 0; k0 < INP; k0 += 32) {
    v16bf bfrag[4];
#pragma unroll
    for (int s = 0; s < 4; ++s)
      bfrag[s] = load_frag(brow[s] + k0, brow[s] + k0 + 8);
#pragma unroll
    for (int mi = 0; mi < 4; ++mi) {
      const v16bf a = load_frag(arow[mi] + k0 + lh * 16,
                                arow[mi] + k0 + lh * 16 + 8);
#pragma unroll
      for (int s = 0; s < 4; ++s) {
        acc[mi][s] = __builtin_amdgcn_wmma_f32_16x16x32_bf16(
            false, a, false, bfrag[s], (short)0, acc[mi][s], false, false);
      }
    }
  }

#pragma unroll
  for (int mi = 0; mi < 4; ++mi)
#pragma unroll
    for (int s = 0; s < 4; ++s)
#pragma unroll
      for (int r = 0; r < 8; ++r)
        XG[(size_t)(m_block + mi * 16 + r + 8 * lh) * G4 +
           ngrp * 64 + s * 16 + lm] = acc[mi][s][r];
}

// ---------------------------------------------------------------
// One LSTM timestep, fused: G = XG_t + h @ Whh^T ; gates ; cell update.
// K=1024 is split in half across wave pairs (256 waves/step instead of 128)
// to halve the WMMA-issue latency on the sequential critical path.
//   wave w in block: m_tile = w&1, k_half = (w>>1)&1, j_sub = w>>2
// k_half==0 seeds its accumulators from the precomputed x-gates;
// k_half==1 accumulates from zero, spills its 4 gate tiles to LDS; after a
// barrier the k_half==0 partner adds them and runs the fused cell update.
// grid = 32 blocks x 256 threads = 256 waves = 2 m x 64 j x 2 k-halves.
__global__ __launch_bounds__(256, 1)
void lstm_step_kernel(const unsigned short* __restrict__ h_in,  // [B, HID] bf16
                      const unsigned short* __restrict__ Whh,   // [4H, HID] bf16
                      const float* __restrict__ xg,             // [B, 4H] f32 (t slice)
                      float* __restrict__ c_st,                 // [B, HID]
                      float* __restrict__ h_f32,                // [B, HID]
                      unsigned short* __restrict__ h_out,       // [B, HID] bf16 (next step)
                      unsigned short* __restrict__ seq_bf,      // [B, HID] bf16 (t slice) or null
                      float* __restrict__ seq_f32) {            // [B, HID] f32 (t slice) or null
  __shared__ float red[4][4][256];     // [pair][gate][tile elem] = 16 KB

  const int lane = threadIdx.x & 31;
  const int w    = threadIdx.x >> 5;   // 0..7
  const int m_tile = w & 1;
  const int k_half = (w >> 1) & 1;
  const int j_sub  = w >> 2;           // 0..1
  const int j_tile = blockIdx.x * 2 + j_sub;
  const int m_base = m_tile * 16;
  const int j_base = j_tile * 16;
  const int lm = lane & 15;
  const int lh = lane >> 4;
  const int col = j_base + lm;
  const int pair = j_sub * 2 + m_tile; // 0..3, shared by the two k-halves

  v8f acc_i, acc_f, acc_g, acc_o;
#pragma unroll
  for (int r = 0; r < 8; ++r) {
    acc_i[r] = 0.0f; acc_f[r] = 0.0f; acc_g[r] = 0.0f; acc_o[r] = 0.0f;
  }
  if (k_half == 0) {
    // Seed from precomputed x-gates (C-tile layout: vgpr r, lane L ->
    // row m_base + r + 8*(L>=16), col = L&15).
#pragma unroll
    for (int r = 0; r < 8; ++r) {
      const float* row = xg + (size_t)(m_base + r + 8 * lh) * G4;
      acc_i[r] = row[col];
      acc_f[r] = row[HID + col];
      acc_g[r] = row[2 * HID + col];
      acc_o[r] = row[3 * HID + col];
    }
  }

  const int kbeg = k_half * (HID / 2);
  const unsigned short* arow = h_in + (size_t)(m_base + lm) * HID;
  const unsigned short* bi = Whh + (size_t)(col)           * HID + lh * 16;
  const unsigned short* bf = Whh + (size_t)(HID + col)     * HID + lh * 16;
  const unsigned short* bg = Whh + (size_t)(2 * HID + col) * HID + lh * 16;
  const unsigned short* bo = Whh + (size_t)(3 * HID + col) * HID + lh * 16;

  for (int k0 = kbeg; k0 < kbeg + HID / 2; k0 += 32) {
    const v16bf a = load_frag(arow + k0 + lh * 16, arow + k0 + lh * 16 + 8);
    v16bf b;
    b = load_frag(bi + k0, bi + k0 + 8);
    acc_i = __builtin_amdgcn_wmma_f32_16x16x32_bf16(false, a, false, b, (short)0, acc_i, false, false);
    b = load_frag(bf + k0, bf + k0 + 8);
    acc_f = __builtin_amdgcn_wmma_f32_16x16x32_bf16(false, a, false, b, (short)0, acc_f, false, false);
    b = load_frag(bg + k0, bg + k0 + 8);
    acc_g = __builtin_amdgcn_wmma_f32_16x16x32_bf16(false, a, false, b, (short)0, acc_g, false, false);
    b = load_frag(bo + k0, bo + k0 + 8);
    acc_o = __builtin_amdgcn_wmma_f32_16x16x32_bf16(false, a, false, b, (short)0, acc_o, false, false);
  }

  // Cross-wave K reduction through LDS (wave-uniform branches; EXEC stays
  // all-ones inside the WMMA loop above).
  if (k_half == 1) {
#pragma unroll
    for (int r = 0; r < 8; ++r) {
      red[pair][0][r * 32 + lane] = acc_i[r];
      red[pair][1][r * 32 + lane] = acc_f[r];
      red[pair][2][r * 32 + lane] = acc_g[r];
      red[pair][3][r * 32 + lane] = acc_o[r];
    }
  }
  __syncthreads();
  if (k_half == 0) {
#pragma unroll
    for (int r = 0; r < 8; ++r) {
      acc_i[r] += red[pair][0][r * 32 + lane];
      acc_f[r] += red[pair][1][r * 32 + lane];
      acc_g[r] += red[pair][2][r * 32 + lane];
      acc_o[r] += red[pair][3][r * 32 + lane];
    }

    // Fused gate nonlinearity + cell update (each lane owns 8 (b,j) cells).
#pragma unroll
    for (int r = 0; r < 8; ++r) {
      const int b = m_base + r + 8 * lh;
      const size_t idx = (size_t)b * HID + col;
      const float iv = sigf(acc_i[r]);
      const float fv = sigf(acc_f[r]);
      const float gv = tanhf(acc_g[r]);
      const float ov = sigf(acc_o[r]);
      const float cn = fv * c_st[idx] + iv * gv;
      const float hn = ov * tanhf(cn);
      c_st[idx] = cn;
      h_f32[idx] = hn;
      const unsigned short hb = f2bf(hn);
      h_out[idx] = hb;
      if (seq_bf)  seq_bf[idx]  = hb;
      if (seq_f32) seq_f32[idx] = hn;
    }
  }
}

// ---------------------------------------------------------------
extern "C" void kernel_launch(void* const* d_in, const int* in_sizes, int n_in,
                              void* d_out, int out_size, void* d_ws, size_t ws_size,
                              hipStream_t stream) {
  (void)in_sizes; (void)n_in; (void)out_size; (void)ws_size;

  const float* x   = (const float*)d_in[0];  // [SEQ, BATCH, INP]
  const float* Wih = (const float*)d_in[1];  // [LAYERS, 4H, INP]
  const float* Whh = (const float*)d_in[2];  // [LAYERS, 4H, HID]
  const float* bih = (const float*)d_in[3];  // [LAYERS, 4H]
  const float* bhh = (const float*)d_in[4];  // [LAYERS, 4H]
  float* out = (float*)d_out;

  // ---- workspace carve-up (~385 MB) ----
  char* ws = (char*)d_ws;
  size_t off = 0;
  auto carve = [&](size_t bytes) -> void* {
    void* p = ws + off;
    off += (bytes + 255) & ~(size_t)255;
    return p;
  };
  const size_t nW   = (size_t)LAYERS * G4 * INP;          // 16.7M per weight set
  const size_t nSeq = (size_t)SEQ * BATCH * HID;          // 16.7M activations
  unsigned short* Wih_bf = (unsigned short*)carve(nW * 2);               // 32 MB
  unsigned short* Whh_bf = (unsigned short*)carve(nW * 2);               // 32 MB
  float*          bias   = (float*)carve((size_t)LAYERS * G4 * 4);       // 64 KB
  unsigned short* seqA   = (unsigned short*)carve(nSeq * 2);             // 32 MB
  unsigned short* seqB   = (unsigned short*)carve(nSeq * 2);             // 32 MB
  float*          XG     = (float*)carve((size_t)SEQ * BATCH * G4 * 4);  // 256 MB
  unsigned short* h_bf0  = (unsigned short*)carve((size_t)BATCH * HID * 2);
  unsigned short* h_bf1  = (unsigned short*)carve((size_t)BATCH * HID * 2);
  float*          h_f32  = (float*)carve((size_t)BATCH * HID * 4);
  float*          c_f32  = (float*)carve((size_t)BATCH * HID * 4);

  // ---- output regions ----
  float* ys = out;                                   // [SEQ, BATCH, HID]
  float* hs = out + nSeq;                            // [LAYERS, BATCH, HID]
  float* cs = hs + (size_t)LAYERS * BATCH * HID;     // [LAYERS, BATCH, HID]

  // ---- one-time conversions (weights stay resident in L2 afterwards) ----
  f32_to_bf16_kernel<<<2048, 256, 0, stream>>>(Wih, Wih_bf, nW);
  f32_to_bf16_kernel<<<2048, 256, 0, stream>>>(Whh, Whh_bf, nW);
  f32_to_bf16_kernel<<<2048, 256, 0, stream>>>(x, seqA, nSeq);
  bias_combine_kernel<<<(LAYERS * G4 + 255) / 256, 256, 0, stream>>>(
      bih, bhh, bias, LAYERS * G4);

  const int nState = BATCH * HID;
  unsigned short* bufs[2] = {seqA, seqB};

  for (int l = 0; l < LAYERS; ++l) {
    const unsigned short* in_bf  = bufs[l & 1];
    unsigned short* out_bf = (l < LAYERS - 1) ? bufs[(l + 1) & 1] : nullptr;
    float* out_f32 = (l == LAYERS - 1) ? ys : nullptr;

    // x-gates for all timesteps: 16384 waves, 64x64 register tiles
    xgate_gemm_kernel<<<2048, 256, 0, stream>>>(
        in_bf, Wih_bf + (size_t)l * G4 * INP, bias + (size_t)l * G4, XG);

    zero_state_kernel<<<(nState + 255) / 256, 256, 0, stream>>>(
        h_bf0, h_bf1, h_f32, c_f32, nState);

    for (int t = 0; t < SEQ; ++t) {
      const unsigned short* hin = (t & 1) ? h_bf1 : h_bf0;
      unsigned short*      hout = (t & 1) ? h_bf0 : h_bf1;
      lstm_step_kernel<<<32, 256, 0, stream>>>(
          hin, Whh_bf + (size_t)l * G4 * HID,
          XG + (size_t)t * BATCH * G4,
          c_f32, h_f32, hout,
          out_bf  ? out_bf  + (size_t)t * BATCH * HID : nullptr,
          out_f32 ? out_f32 + (size_t)t * BATCH * HID : nullptr);
    }

    copy_states_kernel<<<(nState + 255) / 256, 256, 0, stream>>>(
        h_f32, c_f32, hs + (size_t)l * nState, cs + (size_t)l * nState, nState);
  }
}
